// Atomic_GNN_lstm_15522011808196
// MI455X (gfx1250) — compile-verified, matrix operand order checked by reading, not run on hardware
//
#include <hip/hip_runtime.h>
#include <hip/hip_bf16.h>
#include <math.h>

typedef __attribute__((ext_vector_type(16))) _Float16 v16h;
typedef __attribute__((ext_vector_type(8)))  _Float16 v8h;
typedef __attribute__((ext_vector_type(8)))  float    v8f;

#define DEVINL static __device__ __forceinline__

#define S_ 5
#define M_ 4
#define F_ 12

// ---- CDNA5 WMMA fragment helpers (wave32, 16x16x32 f16 -> f32) -------------
// A (16x32 f16): lane L holds row M=L&15; halves 0-7 = K kb..kb+7, halves 8-15 =
//   K 16+kb..16+kb+7, kb = (L&16)?8:0.  B (32x16): same K split, lane = col N.
// => with rows zero-padded to 16/32-half strides, a fragment is 1-2 ds_load_b128.
// C/D (16x16 f32): lane L holds col N=L&15; VGPR r holds row M = r + ((L&16)?8:0).
// Biases are folded into the GEMM: A buffers carry a constant 1.0 at K=kmax and
// the B weight staging places the bias vector in row kmax.

DEVINL int kBase()  { return (threadIdx.x & 16) >> 1; }   // 0 or 8 halves
DEVINL int rowHi()  { return (threadIdx.x & 16) ? 8 : 0; }

DEVINL v8f wmma_f16(v16h a, v16h b, v8f c) {
  return __builtin_amdgcn_wmma_f32_16x16x32_f16(false, a, false, b, (short)0, c, false, false);
}

DEVINL void lds_fence() { asm volatile("s_wait_dscnt 0" ::: "memory"); }

DEVINL v8f zero8() {
  v8f z;
#pragma unroll
  for (int r = 0; r < 8; ++r) z[r] = 0.f;
  return z;
}
DEVINL v8h zeroh8() {
  v8h z;
#pragma unroll
  for (int i = 0; i < 8; ++i) z[i] = (_Float16)0.f;
  return z;
}

// rowPtr: 16B-aligned start of this lane's (zero-padded) row in LDS.
DEVINL v16h fragK16(const _Float16* rowPtr) {           // K <= 16, stride-16 rows
  const v8h lo = *(const v8h*)(rowPtr + kBase());
  const v8h z = zeroh8();
  return __builtin_shufflevector(lo, z, 0,1,2,3,4,5,6,7,8,9,10,11,12,13,14,15);
}
DEVINL v16h fragK32(const _Float16* rowPtr) {           // K <= 32, stride-32 rows
  const v8h lo = *(const v8h*)(rowPtr + kBase());
  const v8h hi = *(const v8h*)(rowPtr + 16 + kBase());
  return __builtin_shufflevector(lo, hi, 0,1,2,3,4,5,6,7,8,9,10,11,12,13,14,15);
}

DEVINL void storeCT(_Float16* buf, int rowStride, int col0, int nValid, v8f acc) {
  const int n = threadIdx.x & 15;
  if (n < nValid) {
#pragma unroll
    for (int r = 0; r < 8; ++r)
      buf[(r + rowHi()) * rowStride + col0 + n] = (_Float16)acc[r];
  }
}

// gfx1250 has a hardware transcendental tanh (v_tanh_f32); fall back to exp.
DEVINL float fast_tanh(float x) {
#if __has_builtin(__builtin_amdgcn_tanhf)
  return __builtin_amdgcn_tanhf(x);
#elif __has_builtin(__builtin_amdgcn_tanh_f32)
  return __builtin_amdgcn_tanh_f32(x);
#else
  const float e = __expf(2.f * x);
  return 1.f - 2.f / (e + 1.f);
#endif
}
DEVINL float fast_sigmoid(float x) { return 1.f / (1.f + __expf(-x)); }

// ============================================================================
// Kernel 1: edge MLP (fc2 folded into msg2) + attention message + GRU step.
// 1 wave/block, 16 samples as WMMA M-dim. Writes lstm_in [N,S,48] f16.
// ============================================================================
__global__ __launch_bounds__(32)
void k1_edge_msg_gru(const float* __restrict__ nodes, const float* __restrict__ pos,
                     const float* __restrict__ att,
                     const float* __restrict__ fc1w, const float* __restrict__ fc1b,
                     const float* __restrict__ fc2w, const float* __restrict__ fc2b,
                     const float* __restrict__ msgw, const float* __restrict__ msgb,
                     const float* __restrict__ msg2w, const float* __restrict__ msg2b,
                     const float* __restrict__ gihw, const float* __restrict__ ghhw,
                     const float* __restrict__ gib, const float* __restrict__ ghb,
                     _Float16* __restrict__ ws)
{
  __shared__ __align__(16) _Float16 sFc1[32 * 32];   // 18x18 (+bias row 18) padded
  __shared__ __align__(16) _Float16 sWc[16 * 32];    // msg2@fc2 12x18 (+bias row 18)
  __shared__ __align__(16) _Float16 sMsg[16 * 16];   // 12x12 (+msg_b row 12)
  __shared__ __align__(16) _Float16 sGih[48 * 32];   // 36x24 (+gib row 24)
  __shared__ __align__(16) _Float16 sGhh[48 * 16];   // 36x12 (+ghb row 12)
  __shared__ __align__(16) _Float16 sNod[16 * 20 * 16]; // [sample][s*4+node][16], col12=1
  __shared__ __align__(16) _Float16 sTh[16 * 20 * 16];
  __shared__ _Float16 sPos[16 * 60];
  __shared__ float    sAtt[16 * 80];
  __shared__ __align__(16) _Float16 sEb[16 * 32];    // e_in rows, col18=1
  __shared__ __align__(16) _Float16 sT[16 * 32];     // tanh(fc1) rows, col18=1
  __shared__ __align__(16) _Float16 sMv[16 * 32];    // m_v rows, col24=1

  const int tid = threadIdx.x;
  const int base = blockIdx.x * 16;
  const int n15 = tid & 15, m16 = tid & 15;
  const int hi = rowHi();

  // ---- stage zero-padded f16 weights with bias folded into row K=kmax ----
  for (int i = tid; i < 32 * 32; i += 32) {
    const int r = i >> 5, c = i & 31;
    _Float16 x = (_Float16)0.f;
    if (r < 18 && c < 18)       x = (_Float16)fc1w[r * 18 + c];
    else if (r < 18 && c == 18) x = (_Float16)fc1b[r];
    sFc1[i] = x;
  }
  for (int i = tid; i < 16 * 32; i += 32) {
    const int r = i >> 5, c = i & 31;
    _Float16 x = (_Float16)0.f;
    if (r < 12 && c < 18)
      x = (_Float16)(msg2w[r * 2] * fc2w[c] + msg2w[r * 2 + 1] * fc2w[18 + c]);
    else if (r < 12 && c == 18)
      x = (_Float16)(msg2w[r * 2] * fc2b[0] + msg2w[r * 2 + 1] * fc2b[1] + msg2b[r]);
    sWc[i] = x;
  }
  for (int i = tid; i < 16 * 16; i += 32) {
    const int r = i >> 4, c = i & 15;
    _Float16 x = (_Float16)0.f;
    if (r < 12 && c < 12)       x = (_Float16)msgw[r * 12 + c];
    else if (r < 12 && c == 12) x = (_Float16)msgb[r];
    sMsg[i] = x;
  }
  for (int i = tid; i < 48 * 32; i += 32) {
    const int r = i >> 5, c = i & 31;
    _Float16 x = (_Float16)0.f;
    if (r < 36 && c < 24)       x = (_Float16)gihw[r * 24 + c];
    else if (r < 36 && c == 24) x = (_Float16)gib[r];
    sGih[i] = x;
  }
  for (int i = tid; i < 48 * 16; i += 32) {
    const int r = i >> 4, c = i & 15;
    _Float16 x = (_Float16)0.f;
    if (r < 36 && c < 12)       x = (_Float16)ghhw[r * 12 + c];
    else if (r < 36 && c == 12) x = (_Float16)ghb[r];
    sGhh[i] = x;
  }

  // ---- stage 16-sample panels, padded per (s,node) slot; col12 = 1.0 ----
  for (int i = tid; i < 16 * 20 * 16; i += 32) {
    const int c = i & 15, sm = i >> 4;
    const int m = sm / 20, slot = sm % 20;
    const float x = (c < 12) ? nodes[(size_t)(base + m) * 240 + slot * 12 + c] : 0.f;
    sNod[i] = (c == 12) ? (_Float16)1.f : (_Float16)x;
    sTh[i]  = (_Float16)fast_tanh(x);
  }
  for (int i = tid; i < 16 * 60; i += 32)
    sPos[i] = (_Float16)pos[(size_t)(base + i / 60) * 60 + (i % 60)];
  for (int i = tid; i < 16 * 80; i += 32)
    sAtt[i] = att[(size_t)(base + i / 80) * 80 + (i % 80)];
  for (int i = tid; i < 16 * 32; i += 32) {
    const int c = i & 31;
    sEb[i] = (c == 18) ? (_Float16)1.f : (_Float16)0.f;
    sT[i]  = (c == 18) ? (_Float16)1.f : (_Float16)0.f;
    sMv[i] = (c == 24) ? (_Float16)1.f : (_Float16)0.f;
  }
  __syncthreads();
  lds_fence();

  // ---- persistent B fragments ----
  const v16h fc1B0 = fragK32(sFc1 + n15 * 32);
  const v16h fc1B1 = fragK32(sFc1 + (16 + n15) * 32);
  const v16h wcB   = fragK32(sWc + n15 * 32);
  const v16h msgB  = fragK16(sMsg + n15 * 16);
  v16h gihB[3], ghhB[3];
#pragma unroll
  for (int g = 0; g < 3; ++g) {
    gihB[g] = fragK32(sGih + (g * 12 + n15) * 32);
    ghhB[g] = fragK16(sGhh + (g * 12 + n15) * 16);
  }

  const int chan = n15;

#pragma unroll 1
  for (int s = 0; s < S_; ++s) {
    // msg_h for all 4 source nodes (kept in accumulator layout)
    v8f mh[M_];
#pragma unroll
    for (int ww = 0; ww < M_; ++ww) {
      const v16h a = fragK16(sNod + (m16 * 20 + s * M_ + ww) * 16);
      v8f acc = wmma_f16(a, msgB, zero8());
#pragma unroll
      for (int r = 0; r < 8; ++r) acc[r] = fmaxf(acc[r], 0.f);
      mh[ww] = acc;
    }

#pragma unroll 1
    for (int v = 0; v < M_; ++v) {
      v8f mv1 = zero8(), mv2 = zero8();
#pragma unroll
      for (int ww = 0; ww < M_; ++ww) {
        const int off = s * 16 + v * 4 + ww;
#pragma unroll
        for (int r = 0; r < 8; ++r) {
          const float a = sAtt[(r + hi) * 80 + off];
          mv1[r] += a * mh[ww][r];
        }
      }
      // e_in rows: [tanh(h_v) | pos_v | pos_w]; cols 0..14 fixed per (s,v)
      for (int i = tid; i < 16 * 15; i += 32) {
        const int m = i / 15, c = i % 15;
        sEb[m * 32 + c] = (c < 12) ? sTh[(m * 20 + s * M_ + v) * 16 + c]
                                   : sPos[m * 60 + (s * M_ + v) * 3 + (c - 12)];
      }
#pragma unroll 1
      for (int ww = 0; ww < M_; ++ww) {
        for (int i = tid; i < 48; i += 32) {          // cols 15..17 = pos_w
          const int m = i / 3, c = i % 3;
          sEb[m * 32 + 15 + c] = sPos[m * 60 + (s * M_ + ww) * 3 + c];
        }
        lds_fence();
        const v16h eA = fragK32(sEb + m16 * 32);
        v8f u0 = wmma_f16(eA, fc1B0, zero8());
        v8f u1 = wmma_f16(eA, fc1B1, zero8());
#pragma unroll
        for (int r = 0; r < 8; ++r) { u0[r] = fast_tanh(u0[r]); u1[r] = fast_tanh(u1[r]); }
        storeCT(sT, 32, 0, 16, u0);                    // C-layout -> padded rows
        storeCT(sT, 32, 16, 2, u1);
        lds_fence();
        const v16h ta = fragK32(sT + m16 * 32);
        v8f me = wmma_f16(ta, wcB, zero8());           // folded fc2 o msg2 (+bias)
        const int off = s * 16 + v * 4 + ww;
#pragma unroll
        for (int r = 0; r < 8; ++r) {
          const float a = sAtt[(r + hi) * 80 + off];
          mv2[r] += a * fmaxf(me[r], 0.f);
        }
      }
      // stage m_v (24-wide) for the GRU A fragment
      if (chan < F_) {
#pragma unroll
        for (int r = 0; r < 8; ++r) {
          sMv[(r + hi) * 32 + chan]      = (_Float16)mv1[r];
          sMv[(r + hi) * 32 + F_ + chan] = (_Float16)mv2[r];
        }
      }
      lds_fence();
      const v16h giA = fragK32(sMv + m16 * 32);
      const v16h ghA = fragK16(sNod + (m16 * 20 + s * M_ + v) * 16);
      v8f gi[3], gh[3];
#pragma unroll
      for (int g = 0; g < 3; ++g) {
        gi[g] = wmma_f16(giA, gihB[g], zero8());
        gh[g] = wmma_f16(ghA, ghhB[g], zero8());
      }
      if (chan < F_) {
#pragma unroll
        for (int r = 0; r < 8; ++r) {
          const float hprev = (float)sNod[((r + hi) * 20 + s * M_ + v) * 16 + chan];
          const float rg = fast_sigmoid(gi[0][r] + gh[0][r]);
          const float zg = fast_sigmoid(gi[1][r] + gh[1][r]);
          const float ng = fast_tanh(gi[2][r] + rg * gh[2][r]);
          const float hnew = (1.f - zg) * ng + zg * hprev;
          // lstm_in layout: [S, F, M] flattened -> col = chan*4 + v
          ws[((size_t)(base + r + hi) * S_ + s) * 48 + chan * M_ + v] = (_Float16)hnew;
        }
      }
    }
  }
}

// ============================================================================
// Kernel 2: forward LSTM scan (5 steps) + single backward step + readout.
// Only hs_b[0] (first backward-scan step, h=c=0) reaches output[:, -1, :].
// ============================================================================
__global__ __launch_bounds__(32)
void k2_bilstm_readout(const _Float16* __restrict__ ws,
                       const float* __restrict__ wihf, const float* __restrict__ whhf,
                       const float* __restrict__ bihf, const float* __restrict__ bhhf,
                       const float* __restrict__ wihb,
                       const float* __restrict__ bihb, const float* __restrict__ bhhb,
                       const float* __restrict__ ro1w, const float* __restrict__ ro1b,
                       const float* __restrict__ ro2w, const float* __restrict__ ro2b,
                       float* __restrict__ out)
{
  __shared__ __align__(16) _Float16 sWihF[104 * 64];  // 96x48 (+bias row 48) padded
  __shared__ __align__(16) _Float16 sWhhF[104 * 32];  // 96x24 padded
  __shared__ __align__(16) _Float16 sWihB[104 * 64];  // 96x48 (+bias row 48) padded
  __shared__ __align__(16) _Float16 sRo1[16 * 64];    // 12x48 (+ro1b row 48)
  __shared__ __align__(16) _Float16 sRo2[16 * 16];    // 6x12 (+ro2b row 12)
  __shared__ __align__(16) _Float16 sX[16 * 5 * 64];  // [sample][t][64], col48=1
  __shared__ __align__(16) _Float16 sH[16 * 32];      // hidden, K=24 pad
  __shared__ float    sG[16 * 96];                    // gate pre-activations
  __shared__ __align__(16) _Float16 sLast[16 * 64];   // [h_f | h_b], col48=1
  __shared__ __align__(16) _Float16 sR[16 * 16];      // relu(ro1), col12=1

  const int tid = threadIdx.x;
  const int base = blockIdx.x * 16;
  const int hi = rowHi(), n15 = tid & 15, m16 = tid & 15;

  for (int i = tid; i < 104 * 64; i += 32) {
    const int r = i >> 6, c = i & 63;
    _Float16 xf = (_Float16)0.f, xb = (_Float16)0.f;
    if (r < 96 && c < 48) {
      xf = (_Float16)wihf[r * 48 + c];
      xb = (_Float16)wihb[r * 48 + c];
    } else if (r < 96 && c == 48) {
      xf = (_Float16)(bihf[r] + bhhf[r]);
      xb = (_Float16)(bihb[r] + bhhb[r]);
    }
    sWihF[i] = xf;
    sWihB[i] = xb;
  }
  for (int i = tid; i < 104 * 32; i += 32) {
    const int r = i >> 5, c = i & 31;
    sWhhF[i] = (r < 96 && c < 24) ? (_Float16)whhf[r * 24 + c] : (_Float16)0.f;
  }
  for (int i = tid; i < 16 * 64; i += 32) {
    const int r = i >> 6, c = i & 63;
    _Float16 x = (_Float16)0.f;
    if (r < 12 && c < 48)       x = (_Float16)ro1w[r * 48 + c];
    else if (r < 12 && c == 48) x = (_Float16)ro1b[r];
    sRo1[i] = x;
    sLast[i] = (c == 48) ? (_Float16)1.f : (_Float16)0.f;
  }
  for (int i = tid; i < 16 * 16; i += 32) {
    const int r = i >> 4, c = i & 15;
    _Float16 x = (_Float16)0.f;
    if (r < 6 && c < 12)       x = (_Float16)ro2w[r * 12 + c];
    else if (r < 6 && c == 12) x = (_Float16)ro2b[r];
    sRo2[i] = x;
    sR[i] = (c == 12) ? (_Float16)1.f : (_Float16)0.f;
  }

  __builtin_prefetch(&ws[(size_t)base * 240], 0, 1);
  for (int i = tid; i < 16 * 320; i += 32) {
    const int m = i / 320, e = i % 320, t = e >> 6, c = e & 63;
    _Float16 x = (_Float16)0.f;
    if (c < 48)       x = ws[(size_t)(base + m) * 240 + t * 48 + c];
    else if (c == 48) x = (_Float16)1.f;
    sX[i] = x;
  }
  for (int i = tid; i < 16 * 32; i += 32) sH[i] = (_Float16)0.f;
  __syncthreads();
  lds_fence();

  v8f c0 = zero8(), c1 = zero8();

#pragma unroll 1
  for (int t = 0; t < S_; ++t) {
    const v16h xA0 = fragK32(sX + m16 * 320 + t * 64);       // K 0..31
    const v16h xA1 = fragK32(sX + m16 * 320 + t * 64 + 32);  // K 32..48 (48 = bias)
    const v16h hA  = fragK32(sH + m16 * 32);                 // K 0..23
#pragma unroll 1
    for (int gt = 0; gt < 8; ++gt) {                // 4 gates x 2 chunks of 24
      const int col0 = (gt >> 1) * 24 + (gt & 1) * 16;
      const int nV = (gt & 1) ? 8 : 16;
      v8f acc = wmma_f16(xA0, fragK32(sWihF + (col0 + n15) * 64), zero8());
      acc = wmma_f16(xA1, fragK32(sWihF + (col0 + n15) * 64 + 32), acc);
      acc = wmma_f16(hA,  fragK32(sWhhF + (col0 + n15) * 32), acc);
      if (n15 < nV) {
#pragma unroll
        for (int r = 0; r < 8; ++r) sG[(r + hi) * 96 + col0 + n15] = acc[r];
      }
    }
    lds_fence();
#pragma unroll
    for (int ct = 0; ct < 2; ++ct) {
      const int chn = ct * 16 + n15;
      if (chn < 24) {
#pragma unroll
        for (int r = 0; r < 8; ++r) {
          const float* gr = &sG[(r + hi) * 96];
          const float ig = fast_sigmoid(gr[chn]);
          const float fg = fast_sigmoid(gr[24 + chn]);
          const float gg = fast_tanh(gr[48 + chn]);
          const float og = fast_sigmoid(gr[72 + chn]);
          float cv = (ct ? c1[r] : c0[r]);
          cv = fg * cv + ig * gg;
          if (ct) c1[r] = cv; else c0[r] = cv;
          sH[(r + hi) * 32 + chn] = (_Float16)(og * fast_tanh(cv));
        }
      }
    }
    lds_fence();
  }
  // forward last hidden -> sLast cols 0..23
  for (int i = tid; i < 16 * 24; i += 32)
    sLast[(i / 24) * 64 + (i % 24)] = sH[(i / 24) * 32 + (i % 24)];
  lds_fence();

  // backward direction: single step on x[S-1] with h=c=0 (no recurrent matmul)
  {
    const v16h xA0 = fragK32(sX + m16 * 320 + 4 * 64);
    const v16h xA1 = fragK32(sX + m16 * 320 + 4 * 64 + 32);
#pragma unroll 1
    for (int gt = 0; gt < 8; ++gt) {
      const int col0 = (gt >> 1) * 24 + (gt & 1) * 16;
      const int nV = (gt & 1) ? 8 : 16;
      v8f acc = wmma_f16(xA0, fragK32(sWihB + (col0 + n15) * 64), zero8());
      acc = wmma_f16(xA1, fragK32(sWihB + (col0 + n15) * 64 + 32), acc);
      if (n15 < nV) {
#pragma unroll
        for (int r = 0; r < 8; ++r) sG[(r + hi) * 96 + col0 + n15] = acc[r];
      }
    }
    lds_fence();
#pragma unroll
    for (int ct = 0; ct < 2; ++ct) {
      const int chn = ct * 16 + n15;
      if (chn < 24) {
#pragma unroll
        for (int r = 0; r < 8; ++r) {
          const float* gr = &sG[(r + hi) * 96];
          const float ig = fast_sigmoid(gr[chn]);
          const float gg = fast_tanh(gr[48 + chn]);
          const float og = fast_sigmoid(gr[72 + chn]);
          const float cv = ig * gg;   // f * c_prev == 0
          sLast[(r + hi) * 64 + 24 + chn] = (_Float16)(og * fast_tanh(cv));
        }
      }
    }
    lds_fence();
  }

  // readout: relu(last @ ro1.T + b1) @ ro2.T + b2
  const v16h lA0 = fragK32(sLast + m16 * 64);
  const v16h lA1 = fragK32(sLast + m16 * 64 + 32);  // includes bias col 48
  v8f r1 = wmma_f16(lA0, fragK32(sRo1 + n15 * 64), zero8());
  r1 = wmma_f16(lA1, fragK32(sRo1 + n15 * 64 + 32), r1);
#pragma unroll
  for (int r = 0; r < 8; ++r) r1[r] = fmaxf(r1[r], 0.f);
  storeCT(sR, 16, 0, 12, r1);
  lds_fence();
  const v16h rA = fragK16(sR + m16 * 16);           // includes bias col 12
  v8f o = wmma_f16(rA, fragK16(sRo2 + n15 * 16), zero8());
  if (n15 < 6) {
#pragma unroll
    for (int r = 0; r < 8; ++r)
      out[(size_t)(base + r + hi) * 6 + n15] = o[r];
  }
}

extern "C" void kernel_launch(void* const* d_in, const int* in_sizes, int n_in,
                              void* d_out, int out_size, void* d_ws, size_t ws_size,
                              hipStream_t stream) {
  const float* nodes = (const float*)d_in[0];
  const float* pos   = (const float*)d_in[1];
  const float* att   = (const float*)d_in[2];
  const float* fc1w  = (const float*)d_in[3];
  const float* fc1b  = (const float*)d_in[4];
  const float* fc2w  = (const float*)d_in[5];
  const float* fc2b  = (const float*)d_in[6];
  const float* msgw  = (const float*)d_in[7];
  const float* msgb  = (const float*)d_in[8];
  const float* msg2w = (const float*)d_in[9];
  const float* msg2b = (const float*)d_in[10];
  const float* gihw  = (const float*)d_in[11];
  const float* ghhw  = (const float*)d_in[12];
  const float* gib   = (const float*)d_in[13];
  const float* ghb   = (const float*)d_in[14];
  const float* wihf  = (const float*)d_in[15];
  const float* whhf  = (const float*)d_in[16];
  const float* bihf  = (const float*)d_in[17];
  const float* bhhf  = (const float*)d_in[18];
  const float* wihb  = (const float*)d_in[19];
  // d_in[20] = lstm_w_hh_b: unused (backward recurrence never reaches output)
  const float* bihb  = (const float*)d_in[21];
  const float* bhhb  = (const float*)d_in[22];
  const float* ro1w  = (const float*)d_in[23];
  const float* ro1b  = (const float*)d_in[24];
  const float* ro2w  = (const float*)d_in[25];
  const float* ro2b  = (const float*)d_in[26];
  float* out = (float*)d_out;
  _Float16* ws16 = (_Float16*)d_ws;   // lstm_in [N, S, 48] f16

  const int N = in_sizes[0] / 240;
  const int blocks = N / 16;          // 16 samples per block (1 wave)

  k1_edge_msg_gru<<<blocks, 32, 0, stream>>>(nodes, pos, att, fc1w, fc1b, fc2w, fc2b,
      msgw, msgb, msg2w, msg2b, gihw, ghhw, gib, ghb, ws16);
  k2_bilstm_readout<<<blocks, 32, 0, stream>>>(ws16, wihf, whhf, bihf, bhhf, wihb,
      bihb, bhhb, ro1w, ro1b, ro2w, ro2b, out);
}
